// BidirectionalSelfAttention_53068615910004
// MI455X (gfx1250) — matrix operand order repaired and checked by reading, not compiled, not run on hardware
//
#include <hip/hip_runtime.h>
#include <hip/hip_bf16.h>

typedef __attribute__((ext_vector_type(16))) __bf16 v16bf;
typedef __attribute__((ext_vector_type(8)))  float  v8f;
typedef __attribute__((ext_vector_type(4)))  unsigned int u32x4;
typedef __attribute__((ext_vector_type(4)))  int i32x4;
typedef __attribute__((ext_vector_type(8)))  int i32x8;

#define BATCH 4
#define TSEQ  2048
#define CDIM  2048
#define NHEAD 16
#define NKV   4
#define HD    128
#define QKVC  3072   // (16 + 2*4) * 128

// ---------- helpers ----------
static __device__ __forceinline__ unsigned short bfbits(float f) {
  __bf16 b = (__bf16)f;
  return __builtin_bit_cast(unsigned short, b);
}
static __device__ __forceinline__ unsigned int packbf(float lo, float hi) {
  return (unsigned int)bfbits(lo) | ((unsigned int)bfbits(hi) << 16);
}
static __device__ __forceinline__ float bflo(unsigned int u) {
  return (float)__builtin_bit_cast(__bf16, (unsigned short)(u & 0xffffu));
}
static __device__ __forceinline__ float bfhi(unsigned int u) {
  return (float)__builtin_bit_cast(__bf16, (unsigned short)(u >> 16));
}

union Frag {
  v16bf v;
  uint4 q[2];       // two 16B chunks -> ds_load_b128 / global_load_b128
  unsigned int u[8];
};

// A-matrix 16x32 bf16 (ISA 7.12.2): lane-half h reads elements
//   u[0..3] = [8h, 8h+8), u[4..7] = [16+8h, 16+8h+8)  -- both 16B contiguous.
static __device__ __forceinline__ Frag load_a_frag(const __bf16* rowp, int half) {
  Frag f;
  f.q[0] = *(const uint4*)(rowp + 8 * half);
  f.q[1] = *(const uint4*)(rowp + 16 + 8 * half);
  return f;
}
// B-matrix 32x16 bf16: lane-half h reads elements [8h, 8h+8) and [16+8h, 16+8h+8) -- same as A.
static __device__ __forceinline__ Frag load_b_frag(const __bf16* colp, int half) {
  Frag f;
  f.q[0] = *(const uint4*)(colp + 8 * half);
  f.q[1] = *(const uint4*)(colp + 16 + 8 * half);
  return f;
}

// ---------------------------------------------------------------------------
// Generic GEMM: Out[M,N] = A[M,K] (f32 or bf16) * W[K,N] (f32), bf16 WMMA.
// Block tile 128x128, BK=32, 256 threads = 8 waves (4 M x 2 N),
// each wave computes 32x64 via 2x4 v_wmma_f32_16x16x32_bf16 accumulators.
// ---------------------------------------------------------------------------
template <bool A_IS_BF16, bool OUT_BF16>
__global__ __launch_bounds__(256)
void gemm_bf16_wmma(const void* __restrict__ Av, const float* __restrict__ W,
                    void* __restrict__ Outv, int M, int N, int K) {
  constexpr int BK  = 32;
  constexpr int LDA = BK + 8;  // 40 bf16 = 80B row stride (16B-aligned, odd*8)
  __shared__ __bf16 As[128 * LDA];  // [m][k]
  __shared__ __bf16 Bs[128 * LDA];  // transposed: [n][k]

  const int tid   = threadIdx.x;
  const int lane  = tid & 31;
  const int wv    = tid >> 5;
  const int half  = lane >> 4;
  const int l16   = lane & 15;
  const int waveM = wv >> 1;  // 0..3
  const int waveN = wv & 1;   // 0..1
  const int bm = blockIdx.y * 128;
  const int bn = blockIdx.x * 128;

  v8f zero = {};
  v8f acc[2][4];
#pragma unroll
  for (int a = 0; a < 2; ++a)
#pragma unroll
    for (int b = 0; b < 4; ++b) acc[a][b] = zero;

  const int ksteps = K / BK;
  for (int kt = 0; kt < ksteps; ++kt) {
    __syncthreads();
    // ---- stage A tile (128 x 32) as bf16 ----
    if (A_IS_BF16) {
      const __bf16* A = (const __bf16*)Av;
#pragma unroll
      for (int i = 0; i < 2; ++i) {
        int idx = tid + i * 256;  // 512 x 16B (4096 bf16)
        int r   = idx >> 2;       // 4 chunks / row
        int c8  = (idx & 3) * 8;
        uint4 val = *(const uint4*)(A + (size_t)(bm + r) * K + kt * BK + c8);
        *(uint4*)(&As[r * LDA + c8]) = val;
      }
    } else {
      const float* A = (const float*)Av;
#pragma unroll
      for (int i = 0; i < 4; ++i) {
        int idx = tid + i * 256;  // 1024 float4
        int r   = idx >> 3;
        int c4  = (idx & 7) * 4;
        const float4 v =
            *(const float4*)(A + (size_t)(bm + r) * K + kt * BK + c4);
        uint2 pk;
        pk.x = packbf(v.x, v.y);
        pk.y = packbf(v.z, v.w);
        *(uint2*)(&As[r * LDA + c4]) = pk;
      }
      if (kt + 1 < ksteps)  // global_prefetch_b8 for next K tile
        __builtin_prefetch(A + (size_t)(bm + (tid >> 1)) * K + (kt + 1) * BK, 0, 1);
    }
    // ---- stage W tile (32 x 128) transposed into Bs[n][k] ----
#pragma unroll
    for (int i = 0; i < 4; ++i) {
      int idx = tid + i * 256;  // 1024 float4
      int k   = idx >> 5;
      int n4  = (idx & 31) * 4;
      const float4 v =
          *(const float4*)(W + (size_t)(kt * BK + k) * N + bn + n4);
      Bs[(n4 + 0) * LDA + k] = (__bf16)v.x;
      Bs[(n4 + 1) * LDA + k] = (__bf16)v.y;
      Bs[(n4 + 2) * LDA + k] = (__bf16)v.z;
      Bs[(n4 + 3) * LDA + k] = (__bf16)v.w;
    }
    __syncthreads();

    Frag bf[4];
#pragma unroll
    for (int nf = 0; nf < 4; ++nf) {
      int n = waveN * 64 + nf * 16 + l16;
      bf[nf] = load_b_frag(&Bs[n * LDA], half);
    }
#pragma unroll
    for (int mf = 0; mf < 2; ++mf) {
      int r = waveM * 32 + mf * 16 + l16;
      Frag af = load_a_frag(&As[r * LDA], half);
#pragma unroll
      for (int nf = 0; nf < 4; ++nf)
        acc[mf][nf] = __builtin_amdgcn_wmma_f32_16x16x32_bf16(
            false, af.v, false, bf[nf].v, (short)0, acc[mf][nf], false, false);
    }
  }

  // ---- epilogue: C layout row = VGPR + 8*half, col = lane&15 ----
#pragma unroll
  for (int mf = 0; mf < 2; ++mf)
#pragma unroll
    for (int nf = 0; nf < 4; ++nf) {
      int col = bn + waveN * 64 + nf * 16 + l16;
#pragma unroll
      for (int i = 0; i < 8; ++i) {
        int row   = bm + waveM * 32 + mf * 16 + i + 8 * half;
        float val = acc[mf][nf][i];
        if (OUT_BF16)
          ((__bf16*)Outv)[(size_t)row * N + col] = (__bf16)val;
        else
          ((float*)Outv)[(size_t)row * N + col] = val;
      }
    }
}

// ---------------------------------------------------------------------------
// RoPE in-place on bf16 qkv: heads 0..15 = Q, 16..19 = K (cols 0..2559).
// ---------------------------------------------------------------------------
__global__ __launch_bounds__(256)
void rope_kernel(__bf16* __restrict__ qkv, const float* __restrict__ freqs) {
  int idx  = blockIdx.x * 256 + threadIdx.x;  // BATCH*TSEQ*20*64 pairs
  int j    = idx & 63;
  int head = (idx >> 6) % 20;
  int bt   = idx / (64 * 20);
  int t    = bt & (TSEQ - 1);
  float c  = freqs[(t * 64 + j) * 2 + 0];
  float s  = freqs[(t * 64 + j) * 2 + 1];
  size_t off = (size_t)bt * QKVC + head * HD + 2 * j;
  unsigned int pu = *(unsigned int*)(qkv + off);
  float e = bflo(pu), o = bfhi(pu);
  *(unsigned int*)(qkv + off) = packbf(e * c - o * s, e * s + o * c);
}

// ---------------------------------------------------------------------------
// Flash attention (bidirectional, GQA). Block = 256 thr = 8 waves; each wave
// owns 16 Q rows (128/block). K tile (via TDM tensor_load_to_lds when
// available) + V^T tile in LDS; Q fragments resident in VGPRs; online softmax
// with 16-lane shfl reductions.
// ---------------------------------------------------------------------------
__global__ __launch_bounds__(256)
void flash_attn_kernel(const __bf16* __restrict__ qkv, __bf16* __restrict__ y) {
  constexpr int BC  = 64;
  constexpr int LDK = HD + 8;  // 136 (272B row; == TDM pad 4 dwords / 64 dwords)
  constexpr int LDV = BC + 8;  // 72  (144B row, 16B-aligned)
  __shared__ __bf16 Ks[BC * LDK];      // [kvrow][d]   (== B^T layout for S)
  __shared__ __bf16 Vt[HD * LDV];      // [d][kvrow]   (== B^T layout for PV)
  __shared__ __bf16 Ps[8 * 16 * LDV];  // per-wave P tile [row][col]

  const int tid  = threadIdx.x;
  const int lane = tid & 31;
  const int wv   = tid >> 5;
  const int half = lane >> 4;
  const int l16  = lane & 15;
  const int b    = blockIdx.z;
  const int h    = blockIdx.y;
  const int kvh  = h >> 2;                  // GQA: 4 Q heads per KV head
  const int qrow0 = blockIdx.x * 128 + wv * 16;
  const float scale = 0.08838834764831845f; // 1/sqrt(128)

  const __bf16* qbase = qkv + (size_t)b * TSEQ * QKVC;

  // Q fragments for this wave's 16 rows (4 K-steps of 32 over HD=128)
  Frag qf[4];
  {
    const __bf16* qp = qbase + (size_t)(qrow0 + l16) * QKVC + h * HD;
#pragma unroll
    for (int kk = 0; kk < 4; ++kk) qf[kk] = load_a_frag(qp + kk * 32, half);
  }

  v8f zero = {};
  v8f accO[8];
#pragma unroll
  for (int nf = 0; nf < 8; ++nf) accO[nf] = zero;
  float m[8], l[8];
#pragma unroll
  for (int i = 0; i < 8; ++i) { m[i] = -3.0e38f; l[i] = 0.f; }

  __bf16* ps = &Ps[wv * 16 * LDV];

  for (int kt = 0; kt < TSEQ / BC; ++kt) {
    const __bf16* kp = qbase + (size_t)(kt * BC) * QKVC + CDIM + kvh * HD;
    const __bf16* vp = kp + NKV * HD;
    __syncthreads();  // previous tile fully consumed

#if __has_builtin(__builtin_amdgcn_tensor_load_to_lds)
    // ---- K tile via Tensor Data Mover: 2D tile 128(d) x 64(rows), bf16,
    //      row stride QKVC, LDS padding 16B per 256B row -> LDK=136 layout.
    if (wv == 0) {
      unsigned long long ga = (unsigned long long)(size_t)kp;
      unsigned int lds_off  = (unsigned int)(size_t)(&Ks[0]);
      u32x4 g0;
      g0[0] = 1u;                                        // count = 1 descriptor
      g0[1] = lds_off;                                   // lds_addr
      g0[2] = (unsigned int)(ga & 0xffffffffu);          // global_addr[31:0]
      g0[3] = (unsigned int)((ga >> 32) & 0x01ffffffu)   // global_addr[56:32]
              | (2u << 30);                              // type = 2 (image)
      i32x8 g1;
      g1[0] = (int)((1u << 16)      // data_size = 2 bytes
                    | (1u << 20)    // pad_enable
                    | (5u << 22)    // pad_interval: 64 DWORDs (256B)
                    | (3u << 25));  // pad_amount: 4 DWORDs (16B)
      g1[1] = (int)(((unsigned)HD & 0xffffu) << 16);     // tensor_dim0 lo
      g1[2] = (int)(((unsigned)BC & 0xffffu) << 16);     // dim0 hi | tensor_dim1
      g1[3] = (int)(((unsigned)HD & 0xffffu) << 16);     // dim1 hi | tile_dim0
      g1[4] = BC;                                        // tile_dim1 | tile_dim2=0
      g1[5] = QKVC;                                      // tensor_dim0_stride lo
      g1[6] = 0;                                         // stride hi | dim1_stride
      g1[7] = 0;
      i32x4 gz4 = {0, 0, 0, 0};
      i32x8 gz8 = {0, 0, 0, 0, 0, 0, 0, 0};
      __builtin_amdgcn_tensor_load_to_lds(g0, g1, gz4, gz4, gz8, 0);
    }
#else
    // ---- K tile manual staging (b128 copies) ----
#pragma unroll
    for (int i = 0; i < 4; ++i) {
      int idx = tid + i * 256;  // 1024 x 16B chunks (64 x 128 bf16)
      int r   = idx >> 4;
      int c8  = (idx & 15) * 8;
      uint4 kw = *(const uint4*)(kp + (size_t)r * QKVC + c8);
      *(uint4*)(&Ks[r * LDK + c8]) = kw;
    }
#endif

    // ---- V tile transposed staging (all waves, overlaps with TDM) ----
#pragma unroll
    for (int i = 0; i < 4; ++i) {
      int idx = tid + i * 256;  // 1024 x 16B chunks (64 x 128 bf16)
      int r   = idx >> 4;
      int c8  = (idx & 15) * 8;
      uint4 vw = *(const uint4*)(vp + (size_t)r * QKVC + c8);
      unsigned int vu[4] = {vw.x, vw.y, vw.z, vw.w};
#pragma unroll
      for (int j = 0; j < 4; ++j) {
        Vt[(c8 + 2 * j + 0) * LDV + r] =
            __builtin_bit_cast(__bf16, (unsigned short)(vu[j] & 0xffffu));
        Vt[(c8 + 2 * j + 1) * LDV + r] =
            __builtin_bit_cast(__bf16, (unsigned short)(vu[j] >> 16));
      }
    }

#if __has_builtin(__builtin_amdgcn_s_wait_tensorcnt) && \
    __has_builtin(__builtin_amdgcn_tensor_load_to_lds)
    if (wv == 0) __builtin_amdgcn_s_wait_tensorcnt((short)0);
#endif
    __syncthreads();  // K (TDM) + V visible to all waves

    // ---- S = Q K^T : 16 x 64, 4 col-frags ----
    v8f accS[4];
#pragma unroll
    for (int nf = 0; nf < 4; ++nf) accS[nf] = zero;
#pragma unroll
    for (int kk = 0; kk < 4; ++kk)
#pragma unroll
      for (int nf = 0; nf < 4; ++nf) {
        int n = nf * 16 + l16;
        Frag bfk = load_b_frag(&Ks[n * LDK + kk * 32], half);
        accS[nf] = __builtin_amdgcn_wmma_f32_16x16x32_bf16(
            false, qf[kk].v, false, bfk.v, (short)0, accS[nf], false, false);
      }

    // ---- online softmax (row = C-frag VGPR index; reduce over 16 lanes) ----
    float mnew[8], alpha[8], rsum[8];
#pragma unroll
    for (int i = 0; i < 8; ++i) {
      float v = accS[0][i];
#pragma unroll
      for (int nf = 1; nf < 4; ++nf) v = fmaxf(v, accS[nf][i]);
      v *= scale;
#pragma unroll
      for (int off = 1; off < 16; off <<= 1) v = fmaxf(v, __shfl_xor(v, off, 32));
      mnew[i]  = fmaxf(m[i], v);
      alpha[i] = __expf(m[i] - mnew[i]);
      rsum[i]  = 0.f;
    }
#pragma unroll
    for (int nf = 0; nf < 4; ++nf)
#pragma unroll
      for (int i = 0; i < 8; ++i) {
        float p = __expf(accS[nf][i] * scale - mnew[i]);
        rsum[i] += p;
        ps[(i + 8 * half) * LDV + nf * 16 + l16] = (__bf16)p;  // per-wave LDS
      }
#pragma unroll
    for (int i = 0; i < 8; ++i) {
      float r = rsum[i];
#pragma unroll
      for (int off = 1; off < 16; off <<= 1) r += __shfl_xor(r, off, 32);
      l[i] = l[i] * alpha[i] + r;
      m[i] = mnew[i];
    }
#pragma unroll
    for (int nf = 0; nf < 8; ++nf)
#pragma unroll
      for (int i = 0; i < 8; ++i) accO[nf][i] *= alpha[i];

    // ---- O += P V : contraction over BC=64 in 2 K-steps ----
#pragma unroll
    for (int kk = 0; kk < 2; ++kk) {
      Frag pf = load_a_frag(&ps[l16 * LDV + kk * 32], half);
#pragma unroll
      for (int nf = 0; nf < 8; ++nf) {
        int n = nf * 16 + l16;
        Frag vf = load_b_frag(&Vt[n * LDV + kk * 32], half);
        accO[nf] = __builtin_amdgcn_wmma_f32_16x16x32_bf16(
            false, pf.v, false, vf.v, (short)0, accO[nf], false, false);
      }
    }
  }

  // ---- epilogue: y = O / l ----
  float inv[8];
#pragma unroll
  for (int i = 0; i < 8; ++i) inv[i] = 1.0f / l[i];
  __bf16* yp = y + ((size_t)b * TSEQ + qrow0) * CDIM + h * HD;
#pragma unroll
  for (int nf = 0; nf < 8; ++nf) {
    int col = nf * 16 + l16;
#pragma unroll
    for (int i = 0; i < 8; ++i)
      yp[(size_t)(i + 8 * half) * CDIM + col] = (__bf16)(accO[nf][i] * inv[i]);
  }
}

// ---------------------------------------------------------------------------
extern "C" void kernel_launch(void* const* d_in, const int* in_sizes, int n_in,
                              void* d_out, int out_size, void* d_ws, size_t ws_size,
                              hipStream_t stream) {
  const float* x      = (const float*)d_in[0];
  const float* freqs  = (const float*)d_in[1];   // [T][64][2] f32
  const float* w_qkv  = (const float*)d_in[2];   // [2048][3072]
  const float* w_proj = (const float*)d_in[3];   // [2048][2048]
  float* out = (float*)d_out;

  const int M = BATCH * TSEQ;  // 8192
  __bf16* qkv_bf = (__bf16*)d_ws;                                  // 48 MB
  __bf16* y_bf   = (__bf16*)((char*)d_ws + (size_t)M * QKVC * 2);  // 32 MB

  // 1) qkv = x @ w_qkv  (f32 -> bf16)
  gemm_bf16_wmma<false, true><<<dim3(QKVC / 128, M / 128), 256, 0, stream>>>(
      x, w_qkv, qkv_bf, M, QKVC, CDIM);
  // 2) RoPE on q,k columns
  rope_kernel<<<(M * 20 * 64) / 256, 256, 0, stream>>>(qkv_bf, freqs);
  // 3) flash attention -> y_bf
  flash_attn_kernel<<<dim3(TSEQ / 128, NHEAD, BATCH), 256, 0, stream>>>(qkv_bf, y_bf);
  // 4) out = y @ w_proj (bf16 -> f32)
  gemm_bf16_wmma<true, false><<<dim3(CDIM / 128, M / 128), 256, 0, stream>>>(
      y_bf, w_proj, out, M, CDIM, CDIM);
}
